// MSDeformAttn_69879117905969
// MI455X (gfx1250) — matrix operand order repaired and checked by reading, not compiled, not run on hardware
//
#include <hip/hip_runtime.h>
#include <hip/hip_bf16.h>

// ---------------------------------------------------------------------------
// MS-Deformable Attention for gfx1250 (MI455X), wave32 + WMMA bf16 split path
// ---------------------------------------------------------------------------

typedef __attribute__((ext_vector_type(16))) __bf16 v16bf;
typedef __attribute__((ext_vector_type(8)))  float  v8f;
typedef __attribute__((ext_vector_type(8)))  unsigned int v8u;

#define B_   2
#define LQ_  21760
#define C_   256
#define NH_  8
#define NL_  4
#define NP_  4
#define HD_  32
#define M_   (B_ * LQ_)          // 43520 rows, divisible by 16
#define KDIM 256
#define LDSLD 260                // 256 + 4 pad -> rows land on distinct banks

__device__ __constant__ int cH[NL_] = {128, 64, 32, 16};
__device__ __constant__ int cW[NL_] = {128, 64, 32, 16};
__device__ __constant__ int cS[NL_] = {0, 16384, 20480, 21504};

// ---------------------------------------------------------------------------
// Weight transpose + bf16 hi/lo split:  W[K][N] f32  ->  WtHi/WtLo[N][K] bf16
// ---------------------------------------------------------------------------
__global__ void msda_wsplit(const float* __restrict__ W,
                            __bf16* __restrict__ hi,
                            __bf16* __restrict__ lo, int N) {
    int t = blockIdx.x * 256 + threadIdx.x;
    if (t >= KDIM * N) return;
    int k = t / N;
    int n = t - k * N;
    float  x  = W[t];
    __bf16 h  = (__bf16)x;
    float  hf = (float)h;
    hi[n * KDIM + k] = h;
    lo[n * KDIM + k] = (__bf16)(x - hf);
}

// ---------------------------------------------------------------------------
// GEMM: C[M][N] = A[M][256] * Wt^T + bias, fp32 via bf16 hi/lo WMMA
// grid.x = M/16, block = 256 threads (8 waves), each wave owns 16-col tiles
// ---------------------------------------------------------------------------
__global__ __launch_bounds__(256) void msda_gemm(
    const float*  __restrict__ A,
    const __bf16* __restrict__ WtHi,
    const __bf16* __restrict__ WtLo,
    const float*  __restrict__ bias,
    float*        __restrict__ Cout,
    int N) {
    __shared__ float As[16 * LDSLD];

    const int row0 = blockIdx.x * 16;
    const int t    = threadIdx.x;

    // Stage 16x256 f32 A tile into LDS (coalesced, 16 loads/thread)
    #pragma unroll
    for (int i = 0; i < 16; ++i) {
        int e = t + i * 256;
        int r = e >> 8, c = e & 255;
        As[r * LDSLD + c] = A[(size_t)(row0 + r) * KDIM + c];
    }
    __syncthreads();

    const int wave = t >> 5;
    const int lane = t & 31;
    const int mrow = lane & 15;
    const int kbA  = (lane >> 4) << 4;   // 0 or 16 (A-fragment K base)
    const int kbB  = (lane >> 4) << 4;   // 0 or 16 (B-fragment K base)
    const int ntiles = N >> 4;

    for (int nt = wave; nt < ntiles; nt += 8) {
        const int ncol = (nt << 4) + (lane & 15);
        v8f acc{};
        #pragma unroll
        for (int kk = 0; kk < 8; ++kk) {
            // ---- A fragment: lane owns row mrow, 16 consecutive K ----
            v16bf ahi, alo;
            const float* ar = &As[mrow * LDSLD + kk * 32 + kbA];
            #pragma unroll
            for (int v2 = 0; v2 < 8; ++v2) {
                int koff = v2 << 1; // {0,2,4,...,14}
                float x0 = ar[koff], x1 = ar[koff + 1];
                __bf16 h0 = (__bf16)x0, h1 = (__bf16)x1;
                ahi[2 * v2]     = h0;
                ahi[2 * v2 + 1] = h1;
                alo[2 * v2]     = (__bf16)(x0 - (float)h0);
                alo[2 * v2 + 1] = (__bf16)(x1 - (float)h1);
            }
            // ---- B fragment: lane owns column ncol, 16 consecutive K ----
            size_t wb = (size_t)ncol * KDIM + kk * 32 + kbB;   // 32B aligned
            v16bf bhi = __builtin_bit_cast(v16bf, *(const v8u*)(WtHi + wb));
            v16bf blo = __builtin_bit_cast(v16bf, *(const v8u*)(WtLo + wb));

            acc = __builtin_amdgcn_wmma_f32_16x16x32_bf16(false, ahi, false, bhi,
                                                          (short)0, acc, false, false);
            acc = __builtin_amdgcn_wmma_f32_16x16x32_bf16(false, ahi, false, blo,
                                                          (short)0, acc, false, false);
            acc = __builtin_amdgcn_wmma_f32_16x16x32_bf16(false, alo, false, bhi,
                                                          (short)0, acc, false, false);
        }
        // C/D layout: VGPR i -> M = i + 8*(lane>=16), N = lane&15
        const float bb   = bias[ncol];
        const int  mbase = (lane >> 4) << 3;
        #pragma unroll
        for (int i = 0; i < 8; ++i)
            Cout[(size_t)(row0 + mbase + i) * N + ncol] = acc[i] + bb;
    }
}

// ---------------------------------------------------------------------------
// Sampling: one wave per (b,h,q). Lane = head channel d (0..31).
// Inline 16-wide softmax via shuffle butterfly, bilinear gather from value.
// ---------------------------------------------------------------------------
__global__ __launch_bounds__(256) void msda_sample(
    const float* __restrict__ value,   // (B, LQ, 256)
    const float* __restrict__ offs,    // (B, LQ, 256)  h*32 + lvl*8 + p*2 + c
    const float* __restrict__ logits,  // (B, LQ, 128)  h*16 + lvl*4 + p
    const float* __restrict__ refpts,  // (B, LQ, NL, 2)
    float*       __restrict__ mid) {   // (B, LQ, 256)
    const int wave = (blockIdx.x << 3) + (threadIdx.x >> 5);
    const int lane = threadIdx.x & 31;
    const int bh = wave & 15;          // B*NH = 16
    const int q  = wave >> 4;
    const int b  = bh >> 3;
    const int h  = bh & 7;
    const size_t bq = (size_t)b * LQ_ + q;

    // ---- softmax over 16 attention logits (both 16-lane halves identical) --
    const int p16 = lane & 15;
    float lg = logits[bq * 128 + h * 16 + p16];
    float mx = lg;
    mx = fmaxf(mx, __shfl_xor(mx, 1, 32));
    mx = fmaxf(mx, __shfl_xor(mx, 2, 32));
    mx = fmaxf(mx, __shfl_xor(mx, 4, 32));
    mx = fmaxf(mx, __shfl_xor(mx, 8, 32));
    float e = __expf(lg - mx);
    float s = e;
    s += __shfl_xor(s, 1, 32);
    s += __shfl_xor(s, 2, 32);
    s += __shfl_xor(s, 4, 32);
    s += __shfl_xor(s, 8, 32);
    const float w = e / s;

    const float* obase = offs + bq * 256 + h * 32;
    const float* rbase = refpts + bq * (NL_ * 2);
    const float* vb    = value + (size_t)b * LQ_ * 256 + h * 32 + lane;

    float acc = 0.f;
    for (int pp = 0; pp < 16; ++pp) {
        const int lvl = pp >> 2, pt = pp & 3;
        const int Hc = cH[lvl], Wc = cW[lvl], st = cS[lvl];
        const float ax = obase[lvl * 8 + pt * 2 + 0];
        const float ay = obase[lvl * 8 + pt * 2 + 1];
        const float lx = rbase[lvl * 2 + 0] + ax / (float)Wc;
        const float ly = rbase[lvl * 2 + 1] + ay / (float)Hc;
        const float x = lx * (float)Wc - 0.5f;
        const float y = ly * (float)Hc - 0.5f;
        const float x0f = floorf(x), y0f = floorf(y);
        const int   x0 = (int)x0f,  y0 = (int)y0f;
        const float fx = x - x0f,   fy = y - y0f;
        const float s00 = (1.f - fx) * (1.f - fy);
        const float s10 = fx * (1.f - fy);
        const float s01 = (1.f - fx) * fy;
        const float s11 = fx * fy;

        float g = 0.f;
        if (x0 >= 0 && x0 < Wc && y0 >= 0 && y0 < Hc)
            g += s00 * vb[(size_t)(st + y0 * Wc + x0) * 256];
        if (x0 + 1 >= 0 && x0 + 1 < Wc && y0 >= 0 && y0 < Hc)
            g += s10 * vb[(size_t)(st + y0 * Wc + x0 + 1) * 256];
        if (x0 >= 0 && x0 < Wc && y0 + 1 >= 0 && y0 + 1 < Hc)
            g += s01 * vb[(size_t)(st + (y0 + 1) * Wc + x0) * 256];
        if (x0 + 1 >= 0 && x0 + 1 < Wc && y0 + 1 >= 0 && y0 + 1 < Hc)
            g += s11 * vb[(size_t)(st + (y0 + 1) * Wc + x0 + 1) * 256];

        acc = fmaf(__shfl(w, pp, 32), g, acc);
    }
    mid[bq * 256 + h * 32 + lane] = acc;
}

// ---------------------------------------------------------------------------
extern "C" void kernel_launch(void* const* d_in, const int* in_sizes, int n_in,
                              void* d_out, int out_size, void* d_ws, size_t ws_size,
                              hipStream_t stream) {
    const float* query  = (const float*)d_in[0];
    const float* refpts = (const float*)d_in[1];
    const float* inpf   = (const float*)d_in[2];
    // d_in[3] (spatial shapes), d_in[4] (level starts): fixed, hardcoded above
    const float* W_off  = (const float*)d_in[5];
    const float* b_off  = (const float*)d_in[6];
    const float* W_attn = (const float*)d_in[7];
    const float* b_attn = (const float*)d_in[8];
    const float* W_val  = (const float*)d_in[9];
    const float* b_val  = (const float*)d_in[10];
    const float* W_out  = (const float*)d_in[11];
    const float* b_out  = (const float*)d_in[12];
    float* out = (float*)d_out;

    // workspace carving (all offsets 256B-aligned)
    char* ws = (char*)d_ws;
    const size_t szMC = (size_t)M_ * 256 * sizeof(float);   // 44.6 MB
    const size_t szMA = (size_t)M_ * 128 * sizeof(float);   // 22.3 MB
    float* value  = (float*)ws;                 ws += szMC;
    float* offbuf = (float*)ws;                 ws += szMC;
    float* attnbf = (float*)ws;                 ws += szMA;
    float* midbuf = (float*)ws;                 ws += szMC;
    __bf16* wtv_hi = (__bf16*)ws;               ws += 256 * 256 * 2;
    __bf16* wtv_lo = (__bf16*)ws;               ws += 256 * 256 * 2;
    __bf16* wto_hi = (__bf16*)ws;               ws += 256 * 256 * 2;
    __bf16* wto_lo = (__bf16*)ws;               ws += 256 * 256 * 2;
    __bf16* wta_hi = (__bf16*)ws;               ws += 256 * 128 * 2;
    __bf16* wta_lo = (__bf16*)ws;               ws += 256 * 128 * 2;
    __bf16* wtw_hi = (__bf16*)ws;               ws += 256 * 256 * 2;
    __bf16* wtw_lo = (__bf16*)ws;               ws += 256 * 256 * 2;

    // 1) weight transpose + bf16 split
    msda_wsplit<<<256, 256, 0, stream>>>(W_val,  wtv_hi, wtv_lo, 256);
    msda_wsplit<<<256, 256, 0, stream>>>(W_off,  wto_hi, wto_lo, 256);
    msda_wsplit<<<128, 256, 0, stream>>>(W_attn, wta_hi, wta_lo, 128);
    msda_wsplit<<<256, 256, 0, stream>>>(W_out,  wtw_hi, wtw_lo, 256);

    const int mb = M_ / 16;   // 2720
    // 2) input projections
    msda_gemm<<<mb, 256, 0, stream>>>(inpf,  wtv_hi, wtv_lo, b_val,  value,  256);
    msda_gemm<<<mb, 256, 0, stream>>>(query, wto_hi, wto_lo, b_off,  offbuf, 256);
    msda_gemm<<<mb, 256, 0, stream>>>(query, wta_hi, wta_lo, b_attn, attnbf, 128);

    // 3) softmax + bilinear sampling: 16 waves per query, 8 waves/block
    msda_sample<<<(B_ * NH_ * LQ_) / 8, 256, 0, stream>>>(value, offbuf, attnbf,
                                                          refpts, midbuf);

    // 4) output projection -> d_out
    msda_gemm<<<mb, 256, 0, stream>>>(midbuf, wtw_hi, wtw_lo, b_out, out, 256);
}
